// Involution2d_37254546326031
// MI455X (gfx1250) — matrix operand order repaired and checked by reading, not compiled, not run on hardware
//
#include <hip/hip_runtime.h>

typedef __attribute__((ext_vector_type(16))) _Float16 v16h;
typedef __attribute__((ext_vector_type(8)))  float    v8f;
typedef __attribute__((ext_vector_type(2)))  float    f32x2;

#define Bn    4
#define Cn    256
#define Hn    64
#define Wn    64
#define HWn   4096
#define NPIX  16384
#define REDn  32
#define Gn    4
#define Kn    7
#define Pn    3
#define KKn   49
#define CGn   64
#define SPANC 196      /* K*K*G */
#define BN_EPS 1e-5f

#define TILE   16
#define HALO   22      /* TILE + Kn - 1 */
#define LPITCH 24      /* padded row pitch: conflict-free for lanes 0-15 / 16-31 */
#define CCH    8       /* channels per LDS chunk */

__device__ __forceinline__ void fill8(v16h& a, int base, const float* __restrict__ p) {
#pragma unroll
  for (int e = 0; e < 8; ++e) a[base + e] = (_Float16)p[e];
}

// -------------------------------------------------------------------------
// K1: red[b,o,hw] = sum_c w_reduce[o,c] * x[b,c,hw]   (M=32, K=256, N=16384)
// One wave per 16-pixel tile; two 16x16 accumulators (o=0..15, 16..31).
// -------------------------------------------------------------------------
__global__ __launch_bounds__(256) void k_reduce_wmma(
    const float* __restrict__ x, const float* __restrict__ w_reduce,
    float* __restrict__ red) {
  const int lane = threadIdx.x & 31;
  const int wave = blockIdx.x * (blockDim.x >> 5) + (threadIdx.x >> 5);
  const int p0   = wave << 4;            // global pixel base (16-aligned)
  const int b    = p0 >> 12;
  const int hw0  = p0 & (HWn - 1);
  const int n    = lane & 15;            // pixel column (B/D) and A row (M)
  const int h    = lane >> 4;

  const float* xb  = x + (size_t)b * Cn * HWn;
  const float* wr0 = w_reduce + n * Cn;          // A rows 0..15
  const float* wr1 = w_reduce + (n + 16) * Cn;   // A rows 16..31

  v8f acc0 = {}; v8f acc1 = {};
  for (int k0 = 0; k0 < Cn; k0 += 32) {
    v16h a0, a1, bt;
    // A layout: element e -> K = (e&7) + 8*h + 16*(e>>3)
    fill8(a0, 0, wr0 + k0 + 8 * h);
    fill8(a0, 8, wr0 + k0 + 16 + 8 * h);
    fill8(a1, 0, wr1 + k0 + 8 * h);
    fill8(a1, 8, wr1 + k0 + 16 + 8 * h);
    // B layout: element e -> K = (e&7) + 8*h + 16*(e>>3), N = lane&15
#pragma unroll
    for (int e = 0; e < 16; ++e)
      bt[e] = (_Float16)xb[(size_t)(k0 + (e & 7) + 8 * h + 16 * (e >> 3)) * HWn + hw0 + n];

    acc0 = __builtin_amdgcn_wmma_f32_16x16x32_f16(false, a0, false, bt,
                                                  (short)0, acc0, false, false);
    acc1 = __builtin_amdgcn_wmma_f32_16x16x32_f16(false, a1, false, bt,
                                                  (short)0, acc1, false, false);
  }
  // D layout: element i -> row = i + 8*h, col = lane&15
  float* rb = red + (size_t)b * REDn * HWn + hw0 + n;
#pragma unroll
  for (int i = 0; i < 8; ++i) {
    const int o = i + 8 * h;
    rb[(size_t)o * HWn]        = acc0[i];
    rb[(size_t)(o + 16) * HWn] = acc1[i];
  }
}

// -------------------------------------------------------------------------
// K2: batch-stat BN -> per-channel scale/shift:  act = relu(red*scale+shift)
// -------------------------------------------------------------------------
__global__ __launch_bounds__(256) void k_stats(
    const float* __restrict__ red, const float* __restrict__ gamma,
    const float* __restrict__ beta, float* __restrict__ stats) {
  __shared__ float s1[256];
  __shared__ float s2[256];
  const int o = blockIdx.x, tid = threadIdx.x;
  float sum = 0.f, sq = 0.f;
  for (int idx = tid; idx < Bn * HWn; idx += 256) {
    const int bb = idx >> 12, hw = idx & (HWn - 1);
    const float v = red[(size_t)(bb * REDn + o) * HWn + hw];
    sum += v; sq += v * v;
  }
  s1[tid] = sum; s2[tid] = sq;
  __syncthreads();
  for (int s = 128; s > 0; s >>= 1) {
    if (tid < s) { s1[tid] += s1[tid + s]; s2[tid] += s2[tid + s]; }
    __syncthreads();
  }
  if (tid == 0) {
    const float inv_n = 1.f / (float)(Bn * HWn);
    const float mean  = s1[0] * inv_n;
    const float var   = s2[0] * inv_n - mean * mean;
    const float scale = gamma[o] * rsqrtf(var + BN_EPS);
    stats[o]        = scale;
    stats[REDn + o] = beta[o] - mean * scale;
  }
}

// -------------------------------------------------------------------------
// K3: ker[b,o,hw] = sum_r w_span[o,r]*relu(bn(red[b,r,hw])) + b_span[o]
// M=196 (13 tiles), K=32 (single WMMA), N=16384. BN+ReLU fused into B tile.
// -------------------------------------------------------------------------
__global__ __launch_bounds__(256) void k_span_wmma(
    const float* __restrict__ red, const float* __restrict__ stats,
    const float* __restrict__ w_span, const float* __restrict__ b_span,
    float* __restrict__ ker) {
  const int lane = threadIdx.x & 31;
  const int wave = blockIdx.x * (blockDim.x >> 5) + (threadIdx.x >> 5);
  const int p0   = wave << 4;
  const int b    = p0 >> 12;
  const int hw0  = p0 & (HWn - 1);
  const int n    = lane & 15;
  const int h    = lane >> 4;

  v16h bt;
  const float* rb = red + (size_t)b * REDn * HWn + hw0 + n;
#pragma unroll
  for (int e = 0; e < 16; ++e) {
    const int r = (e & 7) + 8 * h + 16 * (e >> 3);
    const float v = rb[(size_t)r * HWn] * stats[r] + stats[REDn + r];
    bt[e] = (_Float16)(v > 0.f ? v : 0.f);
  }

  float* kb = ker + (size_t)b * SPANC * HWn + hw0 + n;
  for (int mt = 0; mt < 13; ++mt) {
    const int mbase = mt * 16;
    int row = mbase + n;
    if (row > SPANC - 1) row = SPANC - 1;    // clamp loads in partial tile
    const float* wr = w_span + row * REDn + 8 * h;
    v16h a;
    fill8(a, 0, wr);
    fill8(a, 8, wr + 16);
    v8f acc = {};
    acc = __builtin_amdgcn_wmma_f32_16x16x32_f16(false, a, false, bt,
                                                 (short)0, acc, false, false);
#pragma unroll
    for (int i = 0; i < 8; ++i) {
      const int o = mbase + i + 8 * h;
      if (o < SPANC) kb[(size_t)o * HWn] = acc[i] + b_span[o];
    }
  }
}

// -------------------------------------------------------------------------
// K4: out[b, g*64+cg, h, w] = sum_k ker[b, g*49+k, h, w] * xpad[...]
// 16x16 spatial tile per (b,g); 49 weights in registers; x staged into LDS
// via async global->LDS copies (ASYNCcnt); channel-pair packed-f32 FMAs.
// -------------------------------------------------------------------------
__global__ __launch_bounds__(256) void k_involution(
    const float* __restrict__ x, const float* __restrict__ ker,
    float* __restrict__ out) {
  __shared__ float xs[CCH][HALO][LPITCH];
  const int t  = blockIdx.x;
  const int sp = t & 15;
  const int bg = t >> 4;
  const int g  = bg & 3;
  const int b  = bg >> 2;
  const int h0 = (sp >> 2) << 4;
  const int w0 = (sp & 3) << 4;
  const int tid = threadIdx.x;
  const int px = tid & 15, py = tid >> 4;
  const int hh = h0 + py, ww = w0 + px;
  const int hw = (hh << 6) + ww;

  // per-pixel kernel weights in registers
  float kreg[KKn];
  const float* kp = ker + (size_t)(b * SPANC + g * KKn) * HWn + hw;
#pragma unroll
  for (int k = 0; k < KKn; ++k) kreg[k] = kp[(size_t)k * HWn];

  const float* xg = x   + (size_t)(b * Cn + g * CGn) * HWn;
  float*       og = out + (size_t)(b * Cn + g * CGn) * HWn;

  const unsigned xsBase = (unsigned)(uintptr_t)(&xs[0][0][0]);

  for (int cc = 0; cc < CGn / CCH; ++cc) {
    const float* xc = xg + (size_t)cc * CCH * HWn;
    // cooperative zero-padded halo load: 8 x 22 x 22 elements.
    // In-bounds elements: async global->LDS copy (no VGPR round-trip).
    // Out-of-bounds border: plain DS store of zero.
    for (int idx = tid; idx < CCH * HALO * HALO; idx += 256) {
      const int ch  = idx / (HALO * HALO);
      const int rem = idx - ch * (HALO * HALO);
      const int yy  = rem / HALO;
      const int xx  = rem - yy * HALO;
      const int gh  = h0 + yy - Pn;
      const int gw  = w0 + xx - Pn;
      if ((unsigned)gh < (unsigned)Hn && (unsigned)gw < (unsigned)Wn) {
        const float* gp = xc + (size_t)ch * HWn + (gh << 6) + gw;
#if defined(__gfx1250__)
        const unsigned ldsOff =
            xsBase + (unsigned)(((ch * HALO + yy) * LPITCH + xx) * 4);
        asm volatile("global_load_async_to_lds_b32 %0, %1, off"
                     :: "v"(ldsOff), "v"((unsigned long long)(uintptr_t)gp)
                     : "memory");
#else
        xs[ch][yy][xx] = *gp;
#endif
      } else {
        xs[ch][yy][xx] = 0.f;
      }
    }
#if defined(__gfx1250__)
#if __has_builtin(__builtin_amdgcn_s_wait_asynccnt)
    __builtin_amdgcn_s_wait_asynccnt(0);
#else
    asm volatile("s_wait_asynccnt 0x0" ::: "memory");
#endif
#endif
    __syncthreads();
    if (cc + 1 < CGn / CCH)
      __builtin_prefetch(xc + (size_t)CCH * HWn + hw, 0, 1);
    // channel pairs -> packed f32 FMAs
#pragma unroll
    for (int ch = 0; ch < CCH; ch += 2) {
      f32x2 acc = {0.f, 0.f};
#pragma unroll
      for (int i = 0; i < Kn; ++i)
#pragma unroll
        for (int j = 0; j < Kn; ++j) {
          f32x2 v;
          v.x = xs[ch][py + i][px + j];
          v.y = xs[ch + 1][py + i][px + j];
          acc += v * kreg[i * Kn + j];
        }
      og[(size_t)(cc * CCH + ch) * HWn + hw]     = acc.x;
      og[(size_t)(cc * CCH + ch + 1) * HWn + hw] = acc.y;
    }
    __syncthreads();
  }
}

// -------------------------------------------------------------------------
extern "C" void kernel_launch(void* const* d_in, const int* in_sizes, int n_in,
                              void* d_out, int out_size, void* d_ws, size_t ws_size,
                              hipStream_t stream) {
  (void)in_sizes; (void)n_in; (void)out_size; (void)ws_size;
  const float* x        = (const float*)d_in[0];
  const float* w_reduce = (const float*)d_in[1];
  const float* bn_gamma = (const float*)d_in[2];
  const float* bn_beta  = (const float*)d_in[3];
  const float* w_span   = (const float*)d_in[4];
  const float* b_span   = (const float*)d_in[5];
  float* out = (float*)d_out;

  // workspace layout (floats): red | stats(scale,shift) | ker   (~15 MB)
  float* red   = (float*)d_ws;
  float* stats = red + (size_t)Bn * REDn * HWn;
  float* ker   = stats + 2 * REDn;

  // 1024 wave-tiles of 16 pixels, 8 waves/block -> 128 blocks
  k_reduce_wmma<<<dim3(NPIX / 16 / 8), dim3(256), 0, stream>>>(x, w_reduce, red);
  k_stats      <<<dim3(REDn),          dim3(256), 0, stream>>>(red, bn_gamma, bn_beta, stats);
  k_span_wmma  <<<dim3(NPIX / 16 / 8), dim3(256), 0, stream>>>(red, stats, w_span, b_span, ker);
  k_involution <<<dim3(Bn * Gn * (Hn / TILE) * (Wn / TILE)), dim3(256), 0, stream>>>(x, ker, out);
}
